// PooledMSELoss_26414048870619
// MI455X (gfx1250) — matrix operand order repaired and checked by reading, not compiled, not run on hardware
//
#include <hip/hip_runtime.h>
#include <stdint.h>

// CDNA5 / gfx1250, wave32. Pooled-MSE:
//   out = mean( (boxsum9x9(pred - target) / 81)^2 )
// Separable 9-tap box filter as two banded WMMA matmuls (v_wmma_f32_16x16x32_f16).
// HBM -> LDS staging via global_load_async_to_lds_b32 (ASYNCcnt), guard-free
// operand builds via zero-padded f16 tiles and aligned ds_load_b128.

typedef __attribute__((ext_vector_type(16))) _Float16 v16h;
typedef __attribute__((ext_vector_type(8)))  _Float16 v8h;
typedef __attribute__((ext_vector_type(8)))  float    v8f;
typedef __attribute__((ext_vector_type(4)))  float    v4f;

#define HALO 4
#define TILE 64
#define LIVE 72          // real halo window is 72x72
#define PADR 80          // padded rows/cols so WMMA operand reads need no guards
#define SDS  88          // sD stride in halves (mult of 8 -> 16B-aligned rows)
#define SHS  88          // sHT stride in halves
#define IMG  512
#define NBLK 4096

__global__ __launch_bounds__(256)
void pooled_mse_partial(const float* __restrict__ pred,
                        const float* __restrict__ target,
                        float* __restrict__ partial) {
  // Raw f32 staging for async loads; sHT aliases sP after the diff pass.
  __shared__ __align__(16) unsigned char rawbuf[2 * LIVE * LIVE * 4]; // 41472 B
  __shared__ __align__(16) _Float16 sD[PADR * SDS];                   // 14080 B
  __shared__ float red[256];

  float*    sP  = (float*)rawbuf;
  float*    sT  = (float*)(rawbuf + LIVE * LIVE * 4);
  _Float16* sHT = (_Float16*)rawbuf;   // 64*SHS*2 = 11264 B, reused after barrier

  const int tid  = threadIdx.x;
  const int wave = tid >> 5;
  const int lane = tid & 31;
  const int hsel = (lane >> 4) & 1;    // K-split half per ISA 7.12.2 (wave32)
  const int lc   = lane & 15;

  const int ox = blockIdx.x * TILE;
  const int oy = blockIdx.y * TILE;
  const size_t imgbase = (size_t)blockIdx.z * (IMG * IMG);

  // ---- Phase 1a: async DMA of the live 72x72 windows into LDS (f32) ----
  for (int i = tid; i < LIVE * LIVE; i += 256) {
    int r = i / LIVE, c = i - r * LIVE;
    int gr = oy + r - HALO, gc = ox + c - HALO;
    if ((unsigned)gr < IMG && (unsigned)gc < IMG) {
      const float* gp = pred   + imgbase + (size_t)gr * IMG + gc;
      const float* gt = target + imgbase + (size_t)gr * IMG + gc;
      uint32_t lp = (uint32_t)(uintptr_t)(sP + i);
      uint32_t lt = (uint32_t)(uintptr_t)(sT + i);
      asm volatile("global_load_async_to_lds_b32 %0, %1, off"
                   :: "v"(lp), "v"(gp) : "memory");
      asm volatile("global_load_async_to_lds_b32 %0, %1, off"
                   :: "v"(lt), "v"(gt) : "memory");
    } else {
      sP[i] = 0.0f;
      sT[i] = 0.0f;
    }
  }

  // ---- Phase 1b: zero-fill padded sD (overlaps async latency) ----
  {
    v8h z = {};
    v8h* sd8 = (v8h*)sD;
    for (int i = tid; i < PADR * SDS / 8; i += 256) sd8[i] = z;
  }

  asm volatile("s_wait_asynccnt 0x0" ::: "memory");
  __syncthreads();

  // ---- Phase 1c: d = pred - target, f32 -> f16, 8-wide chunks ----
  for (int i = tid; i < LIVE * (LIVE / 8); i += 256) {   // 72 * 9 = 648 chunks
    int r = i / 9, g = i - r * 9;
    int c0 = g * 8;
    const v4f* p = (const v4f*)(sP + r * LIVE + c0);
    const v4f* t = (const v4f*)(sT + r * LIVE + c0);
    v4f d0 = p[0] - t[0];
    v4f d1 = p[1] - t[1];
    v8h h;
#pragma unroll
    for (int k = 0; k < 4; ++k) {
      h[k]     = (_Float16)d0[k];
      h[4 + k] = (_Float16)d1[k];
    }
    *(v8h*)(sD + r * SDS + c0) = h;
  }
  __syncthreads();   // sP/sT dead from here; sHT takes over that LDS

  // ---- Phase 2: horizontal 9-tap:  H[16x16] = D[16x32] x Bband[32x16] ----
  // Band operand is lane-constant: B[k][c] = (c <= k <= c+8).
  v16h bband;
#pragma unroll
  for (int e = 0; e < 16; ++e) {
    int Kb = hsel ? (e < 8 ? e + 8 : e + 16) : (e < 8 ? e : e + 8); // B layout: lane<16 K in {0..7,16..23}; lane>=16 K in {8..15,24..31}
    bband[e] = (Kb >= lc && Kb <= lc + 8) ? (_Float16)1.0f : (_Float16)0.0f;
  }

  for (int t = wave; t < 20; t += 8) {           // 5 row-tiles x 4 col-tiles
    int rt = t >> 2, ct = t & 3;
    int rb = rt * 16, cb = ct * 16;
    int rowA = rb + lc;                          // A row M = lane&15
    // A f16 16x32 layout: lane<16 K in {0..7,16..23}; lane>=16 K in {8..15,24..31}
    const v8h* a0 = (const v8h*)(sD + rowA * SDS + cb + (hsel ? 8 : 0));
    const v8h* a1 = (const v8h*)(sD + rowA * SDS + cb + (hsel ? 24 : 16));
    v16h a = __builtin_shufflevector(a0[0], a1[0],
                                     0, 1, 2, 3, 4, 5, 6, 7,
                                     8, 9, 10, 11, 12, 13, 14, 15);
    v8f cz = {};
    v8f hacc = __builtin_amdgcn_wmma_f32_16x16x32_f16(false, a, false, bband,
                                                      (short)0, cz, false, false);
    // Store transposed: sHT[col][row]; D VGPR j = rows rb+j (+8 for upper half)
    int col = cb + lc;
    int r0  = rb + (hsel ? 8 : 0);
    v8h hh;
#pragma unroll
    for (int j = 0; j < 8; ++j) hh[j] = (_Float16)hacc[j];
    *(v8h*)(sHT + col * SHS + r0) = hh;
  }
  __syncthreads();

  // ---- Phase 3: vertical 9-tap:  V[16x16] = Aband[16x32] x H[32x16] ----
  v16h aband;
#pragma unroll
  for (int e = 0; e < 16; ++e) {
    int Ka = hsel ? e + 16 : e;
    aband[e] = (Ka >= lc && Ka <= lc + 8) ? (_Float16)1.0f : (_Float16)0.0f;
  }

  float lsum = 0.0f;
  for (int t = wave; t < 16; t += 8) {           // 4x4 output tiles (64x64)
    int rt = t >> 2, ct = t & 3;
    int colB = ct * 16 + lc;                     // B col N = lane&15
    int rB   = rt * 16 + (hsel ? 16 : 0);        // B layout: lane<16 K=0..15, lane>=16 K=16..31
    const v8h* b0 = (const v8h*)(sHT + colB * SHS + rB);
    v16h b = __builtin_shufflevector(b0[0], b0[1],
                                     0, 1, 2, 3, 4, 5, 6, 7,
                                     8, 9, 10, 11, 12, 13, 14, 15);
    v8f cz = {};
    v8f vacc = __builtin_amdgcn_wmma_f32_16x16x32_f16(false, aband, false, b,
                                                      (short)0, cz, false, false);
#pragma unroll
    for (int j = 0; j < 8; ++j) lsum += vacc[j] * vacc[j];
  }

  // ---- deterministic block reduction -> one partial per block ----
  red[tid] = lsum;
  __syncthreads();
  for (int s = 128; s > 0; s >>= 1) {
    if (tid < s) red[tid] += red[tid + s];
    __syncthreads();
  }
  if (tid == 0) {
    int bf = (blockIdx.z * 8 + blockIdx.y) * 8 + blockIdx.x;
    partial[bf] = red[0];
  }
}

__global__ __launch_bounds__(256)
void pooled_mse_finish(const float* __restrict__ partial, float* __restrict__ out) {
  __shared__ float red[256];
  float s = 0.0f;
  for (int i = threadIdx.x; i < NBLK; i += 256) s += partial[i];
  red[threadIdx.x] = s;
  __syncthreads();
  for (int st = 128; st > 0; st >>= 1) {
    if (threadIdx.x < st) red[threadIdx.x] += red[threadIdx.x + st];
    __syncthreads();
  }
  if (threadIdx.x == 0) {
    // mean(diff^2), diff = boxsum/81 ; N = 4*2*8*512*512 = 16777216
    out[0] = red[0] * (float)(1.0 / (6561.0 * 16777216.0));
  }
}

extern "C" void kernel_launch(void* const* d_in, const int* in_sizes, int n_in,
                              void* d_out, int out_size, void* d_ws, size_t ws_size,
                              hipStream_t stream) {
  (void)in_sizes; (void)n_in; (void)out_size; (void)ws_size;
  const float* pred   = (const float*)d_in[0];
  const float* target = (const float*)d_in[1];
  float* out = (float*)d_out;
  float* ws  = (float*)d_ws;   // 4096 floats of scratch partials

  dim3 grid(8, 8, 64);         // 512/64 x 512/64 x (B*C*T)
  pooled_mse_partial<<<grid, 256, 0, stream>>>(pred, target, ws);
  pooled_mse_finish<<<1, 256, 0, stream>>>(ws, out);
}
